// ScaledDotAttention_57664230916119
// MI455X (gfx1250) — matrix operand order repaired and checked by reading, not compiled, not run on hardware
//
#include <hip/hip_runtime.h>
#include <stdint.h>

// ---- problem constants (fixed by the reference) ----
constexpr int B = 2, L = 2048, S = 2048, H = 8, E = 64;
constexpr float SCALE = 0.125f;        // 1/sqrt(64)
constexpr float EPS   = 1e-8f;
constexpr size_t V_SZ = (size_t)B * L * H * E;   // output 0
constexpr size_t A_SZ = (size_t)B * H * L * S;   // output 1
constexpr int MT = L / 16;             // 128 query tiles per (b,h)

typedef __attribute__((ext_vector_type(16))) _Float16 v16h;
typedef __attribute__((ext_vector_type(8)))  float    v8f;
typedef __attribute__((ext_vector_type(4)))  float    v4f;

// =====================================================================
// WMMA register layouts (ISA 7.12.2, wave32):
//   A 16x32 f16 : lane l -> M=l%16, elem e -> K = (e/8)*16 + (l/16)*8 + e%8
//   B 32x16 f16 : lane l -> N=l%16, elem e -> K = (l/16)*16 + e (not ISA order; used here for A and B)
//   C/D 16x16 f32: lane l, vgpr v -> M = v + 8*(l/16), N = l%16
// =====================================================================

__device__ __forceinline__ void load_b_frags(const float* __restrict__ krow,
                                             int half, v16h& b0, v16h& b1) {
#pragma unroll
  for (int f = 0; f < 2; ++f) {
    const float* p = krow + f * 32 + half * 16;
    v4f y0 = *(const v4f*)p;       v4f y1 = *(const v4f*)(p + 4);
    v4f y2 = *(const v4f*)(p + 8); v4f y3 = *(const v4f*)(p + 12);
#pragma unroll
    for (int j = 0; j < 4; ++j) {
      if (f == 0) { b0[j]   = (_Float16)y0[j]; b0[4+j]  = (_Float16)y1[j];
                    b0[8+j] = (_Float16)y2[j]; b0[12+j] = (_Float16)y3[j]; }
      else        { b1[j]   = (_Float16)y0[j]; b1[4+j]  = (_Float16)y1[j];
                    b1[8+j] = (_Float16)y2[j]; b1[12+j] = (_Float16)y3[j]; }
    }
  }
}

// One 16x16 score tile: WMMA + scale + masks + scattered store.
// DIAG=true additionally applies the per-element causal test using pos.
template <bool DIAG>
__device__ __forceinline__ void qk_tile(
    const v16h& a0, const v16h& a1, const float* __restrict__ Kd,
    const unsigned char* __restrict__ mk, const int* __restrict__ pos,
    float* __restrict__ arow, int b, int h, int nBase, int mBase,
    int half, int lm) {
  const float* krow = Kd + (((size_t)b * S + nBase + lm) * H + h) * E;
  v16h b0, b1;
  load_b_frags(krow, half, b0, b1);

  v8f c = {};
  c = __builtin_amdgcn_wmma_f32_16x16x32_f16(false, a0, false, b0, (short)0, c, false, false);
  c = __builtin_amdgcn_wmma_f32_16x16x32_f16(false, a1, false, b1, (short)0, c, false, false);

  const int  ng    = nBase + lm;
  const bool kmiss = mk[(size_t)b * S + ng] != 0;
  int pk = 0;
  if (DIAG) pk = pos[b * L + ng];            // S == L

#pragma unroll
  for (int v = 0; v < 8; ++v) {
    const int M = v + half * 8;
    float val = c[v] * SCALE;
    bool msk = kmiss;
    if (DIAG) {
      const int pq = pos[b * L + mBase + M];
      msk |= (pk > pq);
    }
    arow[(size_t)M * S + ng] = msk ? -INFINITY : val;
  }
}

// =====================================================================
// Kernel 1: scores = scale * (Q K^T) with key-miss / causal masking.
// One wave per 16-row Q strip per (b,h). Main loop is branch-free; the
// diagonal tile (only one that needs per-element causal) is peeled.
// =====================================================================
__global__ __launch_bounds__(128) void qk_kernel(
    const float* __restrict__ Q, const float* __restrict__ Kd,
    const unsigned char* __restrict__ mk, const int* __restrict__ pos,
    const int* __restrict__ cz, float* __restrict__ A)
{
  const int wid   = blockIdx.x * 4 + (threadIdx.x >> 5);
  const int lane  = threadIdx.x & 31;
  const int mTile = wid & (MT - 1);
  const int bh    = wid >> 7;              // MT == 128
  const int h     = bh & (H - 1);
  const int b     = bh >> 3;
  const int mBase = mTile * 16;
  const int half  = lane >> 4;
  const int lm    = lane & 15;
  const int causal = cz[0];

  // ---- load Q strip into two A-fragments (E = 0..31, 32..63) ----
  v16h a0, a1;
  const float* qrow = Q + (((size_t)b * L + mBase + lm) * H + h) * E;
#pragma unroll
  for (int f = 0; f < 2; ++f) {
#pragma unroll
    for (int r = 0; r < 2; ++r) {
      const float* p = qrow + f * 32 + half * 16 + r * 8;
      v4f x0 = *(const v4f*)p;
      v4f x1 = *(const v4f*)(p + 4);
#pragma unroll
      for (int j = 0; j < 4; ++j) {
        if (f == 0) { a0[r*8 + j] = (_Float16)x0[j]; a0[r*8 + 4 + j] = (_Float16)x1[j]; }
        else        { a1[r*8 + j] = (_Float16)x0[j]; a1[r*8 + 4 + j] = (_Float16)x1[j]; }
      }
    }
  }

  float* arow = A + ((size_t)bh * L + mBase) * S;

  // Full (never causally masked) tiles: nt < mTile when causal, else all.
  const int nFull = causal ? mTile : (S / 16);
  for (int nt = 0; nt < nFull; ++nt) {
    // prefetch next K tile row (gfx1250 global_prefetch_b8)
    if (nt + 1 < nFull) {
      const float* nxt = Kd + (((size_t)b * S + (nt + 1) * 16 + lm) * H + h) * E + half * 16;
      __builtin_prefetch(nxt, 0, 3);
    }
    qk_tile<false>(a0, a1, Kd, mk, pos, arow, b, h, nt * 16, mBase, half, lm);
  }
  // Peeled diagonal tile (per-element causal via pos).
  if (causal) {
    qk_tile<true>(a0, a1, Kd, mk, pos, arow, b, h, mTile * 16, mBase, half, lm);
  }
}

// =====================================================================
// Kernel 2: per-row softmax + q-miss relu zeroing + entropy.
// One 256-thread block per row. Overwrites every A element, so
// causal-masked positions become exact zeros (K3 relies on this).
// =====================================================================
__device__ __forceinline__ float warpRedMax(float v) {
#pragma unroll
  for (int m = 16; m > 0; m >>= 1) v = fmaxf(v, __shfl_xor(v, m, 32));
  return v;
}
__device__ __forceinline__ float warpRedSum(float v) {
#pragma unroll
  for (int m = 16; m > 0; m >>= 1) v += __shfl_xor(v, m, 32);
  return v;
}

__global__ __launch_bounds__(256) void softmax_kernel(
    const unsigned char* __restrict__ mq, const int* __restrict__ pos,
    const int* __restrict__ cz, float* __restrict__ A, float* __restrict__ Ent)
{
  __shared__ float sm[8];
  const int row = blockIdx.x;                  // (b*H + h)*L + l
  const int l   = row & (L - 1);
  const int b   = row / (H * L);
  float* rp = A + (size_t)row * S;
  const int  causal = cz[0];
  const int  pi     = pos[b * L + l];
  const bool qmiss  = mq[(size_t)b * L + l] != 0;
  const int  t      = threadIdx.x;
  const int  w      = t >> 5;

  float vals[S / 256];
  float mx = -INFINITY;
#pragma unroll
  for (int i = 0; i < S / 256; ++i) {
    const int j = t + i * 256;
    const bool masked = causal && (pos[b * L + j] > pi);
    const float s = masked ? -INFINITY : rp[j];
    vals[i] = s;
    mx = fmaxf(mx, s);
  }
  // block max
  mx = warpRedMax(mx);
  if ((t & 31) == 0) sm[w] = mx;
  __syncthreads();
  if (t < 32) {
    float x = (t < 8) ? sm[t] : -INFINITY;
    x = warpRedMax(x);
    if (t == 0) sm[0] = x;
  }
  __syncthreads();
  mx = sm[0];
  __syncthreads();

  float sum = 0.f;
#pragma unroll
  for (int i = 0; i < S / 256; ++i) {
    const float p = (vals[i] == -INFINITY) ? 0.f : __expf(vals[i] - mx);
    vals[i] = p;
    sum += p;
  }
  sum = warpRedSum(sum);
  if ((t & 31) == 0) sm[w] = sum;
  __syncthreads();
  if (t < 32) {
    float x = (t < 8) ? sm[t] : 0.f;
    x = warpRedSum(x);
    if (t == 0) sm[0] = x;
  }
  __syncthreads();
  const float inv = 1.f / sm[0];
  __syncthreads();

  float ent = 0.f;
#pragma unroll
  for (int i = 0; i < S / 256; ++i) {
    const float att = qmiss ? 0.f : vals[i] * inv;   // relu(softmax + m_q)
    rp[t + i * 256] = att;
    ent -= att * __logf(fmaxf(att, EPS));            // att==0 -> 0 contribution
  }
  ent = warpRedSum(ent);
  if ((t & 31) == 0) sm[w] = ent;
  __syncthreads();
  if (t == 0) {
    float e = 0.f;
#pragma unroll
    for (int i = 0; i < 8; ++i) e += sm[i];
    Ent[row] = e;
  }
}

// =====================================================================
// Kernel 3: V = A @ value, contraction bounded by causal diagonal.
// One wave per 16-row strip; value chunks staged through LDS.
// =====================================================================
__global__ __launch_bounds__(32) void av_kernel(
    const float* __restrict__ Val, const int* __restrict__ cz,
    const float* __restrict__ A, float* __restrict__ Out)
{
  __shared__ float lv[32 * 64];                 // 8 KB value staging tile
  const int wid   = blockIdx.x;
  const int lane  = threadIdx.x;
  const int mTile = wid & (MT - 1);
  const int bh    = wid >> 7;
  const int h     = bh & (H - 1);
  const int b     = bh >> 3;
  const int mBase = mTile * 16;
  const int half  = lane >> 4;
  const int lm    = lane & 15;
  const int causal = cz[0];
  const int kEnd   = causal ? ((mBase + 16 + 31) & ~31) : S;  // A==0 past diag

  v8f cacc[4] = { {}, {}, {}, {} };            // 16x64 f32 accumulators
  const float* arow = A + ((size_t)bh * L + mBase + lm) * S;

  for (int kb = 0; kb < kEnd; kb += 32) {
    // ---- A fragment (f32 -> f16), K = kb..kb+31 ----
    v16h a;
#pragma unroll
    for (int r = 0; r < 2; ++r) {
      const float* p = arow + kb + half * 16 + r * 8;
      v4f x0 = *(const v4f*)p;
      v4f x1 = *(const v4f*)(p + 4);
#pragma unroll
      for (int j = 0; j < 4; ++j) {
        a[r*8 + j] = (_Float16)x0[j];
        a[r*8 + 4 + j] = (_Float16)x1[j];
      }
    }
    // prefetch next value chunk
    if (kb + 32 < kEnd) {
      __builtin_prefetch(Val + (((size_t)b * S + kb + 32 + half) * H + h) * E + lm * 4, 0, 3);
    }
    // ---- stage value rows kb..kb+31 (all 64 cols) into LDS, coalesced ----
#pragma unroll
    for (int i = 0; i < 16; ++i) {
      const int sl = 2 * i + half;             // local row 0..31
      const int e0 = lm * 4;                   // col 0..60
      v4f x = *(const v4f*)(Val + (((size_t)b * S + kb + sl) * H + h) * E + e0);
      *(v4f*)(&lv[sl * 64 + e0]) = x;
    }
    __syncthreads();
    // ---- 4 B-fragments (one per 16-wide E tile) + WMMA ----
#pragma unroll
    for (int t = 0; t < 4; ++t) {
      v16h bb;
#pragma unroll
      for (int e = 0; e < 16; ++e)
        bb[e] = (_Float16)lv[(half * 16 + e) * 64 + t * 16 + lm];
      cacc[t] = __builtin_amdgcn_wmma_f32_16x16x32_f16(false, a, false, bb,
                                                       (short)0, cacc[t], false, false);
    }
    __syncthreads();
  }

  // ---- store V tile (B,L,H,E) ----
#pragma unroll
  for (int t = 0; t < 4; ++t) {
#pragma unroll
    for (int v = 0; v < 8; ++v) {
      const int M = v + half * 8;
      Out[(((size_t)b * L + mBase + M) * H + h) * E + t * 16 + lm] = cacc[t][v];
    }
  }
}

// =====================================================================
extern "C" void kernel_launch(void* const* d_in, const int* in_sizes, int n_in,
                              void* d_out, int out_size, void* d_ws, size_t ws_size,
                              hipStream_t stream) {
  const float*         q   = (const float*)d_in[0];
  const float*         k   = (const float*)d_in[1];
  const float*         v   = (const float*)d_in[2];
  const unsigned char* mk  = (const unsigned char*)d_in[3];  // bool (B,S,1)
  const unsigned char* mq  = (const unsigned char*)d_in[4];  // bool (B,L,1)
  const int*           pos = (const int*)d_in[5];
  const int*           cz  = (const int*)d_in[6];

  float* out  = (float*)d_out;
  float* Vout = out;
  float* Aout = out + V_SZ;
  float* Eout = Aout + A_SZ;

  const int nStrips = B * H * MT;                       // 2048 waves
  qk_kernel<<<nStrips / 4, 128, 0, stream>>>(q, k, mk, pos, cz, Aout);
  softmax_kernel<<<B * H * L, 256, 0, stream>>>(mq, pos, cz, Aout, Eout);
  av_kernel<<<nStrips, 32, 0, stream>>>(v, cz, Aout, Vout);
}
